// BasicBlock_bi_1632087573067
// MI455X (gfx1250) — matrix operand order repaired and checked by reading, not compiled, not run on hardware
//
#include <hip/hip_runtime.h>

// MI455X / gfx1250, wave32. bf16 WMMA implicit-GEMM ResNet BasicBlock.
// conv1: stages x (NCHW f32 -> bf16 LDS tile), WMMA, writes y1 NHWC bf16 + BN1 stats.
// conv2: BN1 folded into weights (w2*a1, per-tap bias from b1); stages y1 via
//        global_load_async_to_lds_b128 (ASYNCcnt), WMMA, +residual, BN2 stats.
//
// Workspace layout (bytes), total ~67.3 MB:
//   0      : wbf1  (64*576 bf16 = 73728)   repacked conv1 weights [oc][tap][ic]
//   73728  : wbf2  (73728)                 conv2 weights folded with BN1 scale
//   147456 : stats (256 f32)               sum1[64],sumsq1[64],sum2[64],sumsq2[64]
//   148480 : params(256 f32)               a1[64],b1[64],a2[64],b2[64]
//   149504 : btap  (576 f32 = 2304)        per-tap BN1-shift bias for conv2
//   151808 : y1    (32*128*128*64 bf16 = 67108864)  conv1 output, NHWC

typedef __attribute__((ext_vector_type(16))) __bf16 v16bf;
typedef __attribute__((ext_vector_type(8)))  __bf16 v8bf;
typedef __attribute__((ext_vector_type(8)))  float  v8f;
typedef __attribute__((address_space(3)))    unsigned short lds_u16;

static constexpr int Nn = 32, Cc = 64, Hh = 128, Ww = 128;
static constexpr int Kred = 576;              // 9 taps * 64 ic
static constexpr float CNT = (float)Nn * Hh * Ww;

__device__ __forceinline__ v16bf cat16(v8bf lo, v8bf hi) {
  return __builtin_shufflevector(lo, hi, 0,1,2,3,4,5,6,7,8,9,10,11,12,13,14,15);
}

// ---- repack conv1 weights [O][I][3][3] f32 -> [O][tap][ic] bf16 ----------
__global__ void prep_w(const float* __restrict__ w, unsigned short* __restrict__ wbraw) {
  int i = blockIdx.x * 256 + threadIdx.x;
  if (i < Cc * Kred) {
    int o = i / Kred, r = i % Kred;
    int tap = r >> 6, ic = r & 63;
    ((__bf16*)wbraw)[i] = (__bf16)w[(o * 64 + ic) * 9 + tap];
  }
}

// ---- fold BN1 affine into conv2 weights + per-tap bias -------------------
// wbf2[oc][tap][ic] = bf16(w2[oc][ic][tap] * a1[ic])
// btap[oc][tap]     = sum_ic w2[oc][ic][tap] * b1[ic]
__global__ void fold_w2(const float* __restrict__ w2, const float* __restrict__ a1,
                        const float* __restrict__ b1, unsigned short* __restrict__ wbraw,
                        float* __restrict__ btap) {
  int tap = blockIdx.x;      // 0..8
  int oc  = threadIdx.x;     // 0..63
  float s = 0.0f;
  for (int ic = 0; ic < 64; ++ic) {
    float w = w2[(oc * 64 + ic) * 9 + tap];
    ((__bf16*)wbraw)[oc * Kred + tap * 64 + ic] = (__bf16)(w * a1[ic]);
    s += w * b1[ic];
  }
  btap[oc * 9 + tap] = s;
}

__global__ void zero_f(float* __restrict__ p, int n) {
  int i = blockIdx.x * 256 + threadIdx.x;
  if (i < n) p[i] = 0.0f;
}

// ---- implicit-GEMM 3x3 conv via v_wmma_f32_16x16x32_bf16 -----------------
// Block: 256 threads = 8 waves; tile = 64 oc x 32 pixels (one image row seg).
template <bool IS2>
__global__ __launch_bounds__(256)
void conv_bb(const float* __restrict__ x,            // original input, NCHW f32
             const unsigned short* __restrict__ wraw,// repacked weights bf16
             const unsigned short* __restrict__ y1in,// pass2: y1 NHWC bf16
             unsigned short* __restrict__ y1out,     // pass1: y1 NHWC bf16
             float* __restrict__ zout,               // pass2: z NCHW f32 (d_out)
             const float* __restrict__ btap,         // pass2: per-tap bias
             float* __restrict__ gsum, float* __restrict__ gsq) {
  __shared__ __bf16 ilds[3 * 34 * 64];   // halo tile [row][col][ic], 13056 B
  __shared__ float  ssum[64], ssq[64];
  __shared__ float  sBt[576];            // per-tap bias (pass2)

  const int tid  = threadIdx.x;
  const int lane = tid & 31;
  const int wave = tid >> 5;
  const int n  = blockIdx.z;
  const int h  = blockIdx.y;
  const int wb = blockIdx.x * 32;
  const __bf16* wbf = (const __bf16*)wraw;

  if (tid < 64) { ssum[tid] = 0.0f; ssq[tid] = 0.0f; }

  // -------- stage input halo tile (3 rows x 34 cols x 64 ic) to LDS ------
  if constexpr (!IS2) {
    // x NCHW f32: thread handles one (row, ic), walks 34 cols; convert->bf16
    for (int p = tid; p < 3 * 64; p += 256) {
      int row = p >> 6, ic = p & 63;
      int hh = h + row - 1;
      bool rok = (hh >= 0) && (hh < Hh);
      const float* src = x + (((size_t)n * Cc + ic) * Hh + (rok ? hh : 0)) * Ww;
      __bf16* dst = ilds + row * 34 * 64 + ic;
      for (int col = 0; col < 34; ++col) {
        int ww = wb + col - 1;
        float v = (rok && ww >= 0 && ww < Ww) ? src[ww] : 0.0f;
        dst[col * 64] = (__bf16)v;
      }
    }
  } else {
    for (int p = tid; p < 576; p += 256) sBt[p] = btap[p];
    // y1 NHWC bf16: each in-bounds (row,col) is a contiguous 128B run ->
    // async copy straight into LDS (gfx1250 data mover path, ASYNCcnt).
    for (int p = tid; p < 3 * 34; p += 256) {
      int row = p / 34, col = p % 34;
      int hh = h + row - 1, ww = wb + col - 1;
      __bf16* dst = ilds + (row * 34 + col) * 64;
      if (hh >= 0 && hh < Hh && ww >= 0 && ww < Ww) {
        const __bf16* src = (const __bf16*)y1in + (((size_t)n * Hh + hh) * Ww + ww) * 64;
#pragma unroll
        for (int j = 0; j < 8; ++j) {
          lds_u16* d3 = (lds_u16*)(dst + j * 8);
          asm volatile("global_load_async_to_lds_b128 %0, %1, off"
                       :: "v"(d3), "v"(src + j * 8) : "memory");
        }
      } else {
        v8bf z = {};
#pragma unroll
        for (int j = 0; j < 8; ++j) *(v8bf*)(dst + j * 8) = z;
      }
    }
    asm volatile("s_wait_asynccnt 0x0" ::: "memory");
  }
  __syncthreads();

  // -------- K loop: 9 taps x 2 chunks of 32 ic, one WMMA each ------------
  const int mbase = (wave >> 1) * 16;   // oc tile base
  const int ntile = wave & 1;           // pixel tile
  const int mrow  = lane & 15;          // A row / B column
  const int khalf = lane >> 4;          // K-half select (this code's contiguous k = 16h + s)

  v8f acc = {};
  const __bf16* arow = wbf + (size_t)(mbase + mrow) * Kred + khalf * 16;
#pragma unroll
  for (int t = 0; t < 9; ++t) {
    const int kh = t / 3, kw = t % 3;
    const __bf16* bcol = ilds + (kh * 34 + (ntile * 16 + mrow + kw)) * 64 + khalf * 16;
#pragma unroll
    for (int c = 0; c < 2; ++c) {
      const v8bf* pa = (const v8bf*)(arow + t * 64 + c * 32);
      v16bf A = cat16(pa[0], pa[1]);              // K khalf*16 + 0..15
      const v8bf* pb = (const v8bf*)(bcol + c * 32);
      v16bf B = cat16(pb[0], pb[1]);              // K khalf*16 + 0..15
      acc = __builtin_amdgcn_wmma_f32_16x16x32_bf16(
          false, A, false, B, (short)0, acc, false, false);
    }
  }

  // -------- epilogue ------------------------------------------------------
  const int px   = ntile * 16 + mrow;
  const int wpix = wb + px;
  const int ocb  = mbase + khalf * 8;   // C layout: VGPR r -> oc = ocb + r

  if constexpr (!IS2) {
    v8bf pk;
#pragma unroll
    for (int r = 0; r < 8; ++r) pk[r] = (__bf16)acc[r];
    *(v8bf*)((__bf16*)y1out + (((size_t)n * Hh + h) * Ww + wpix) * 64 + ocb) = pk;
#pragma unroll
    for (int r = 0; r < 8; ++r) {
      float v = acc[r];
      atomicAdd(&ssum[ocb + r], v);
      atomicAdd(&ssq[ocb + r], v * v);
    }
  } else {
#pragma unroll
    for (int r = 0; r < 8; ++r) {
      int oc = ocb + r;
      // bias over in-bounds taps only (matches zero-padding of BN1 output)
      float bias = 0.0f;
#pragma unroll
      for (int kh = 0; kh < 3; ++kh) {
        int hh = h + kh - 1;
        if (hh < 0 || hh >= Hh) continue;
#pragma unroll
        for (int kw = 0; kw < 3; ++kw) {
          int wwp = wpix + kw - 1;
          if (wwp < 0 || wwp >= Ww) continue;
          bias += sBt[oc * 9 + kh * 3 + kw];
        }
      }
      size_t gi = (((size_t)n * Cc + oc) * Hh + h) * Ww + wpix;
      float v = acc[r] + bias + x[gi];   // conv2 + folded BN1 shift + residual
      zout[gi] = v;
      atomicAdd(&ssum[oc], v);
      atomicAdd(&ssq[oc], v * v);
    }
  }
  __syncthreads();
  if (tid < 64) {
    atomicAdd(&gsum[tid], ssum[tid]);
    atomicAdd(&gsq[tid], ssq[tid]);
  }
}

// ---- fold BN stats into affine a,b ---------------------------------------
__global__ void bn_params(const float* __restrict__ sum, const float* __restrict__ sumsq,
                          const float* __restrict__ gamma, const float* __restrict__ beta,
                          float* __restrict__ a, float* __restrict__ b) {
  int c = threadIdx.x;
  if (c < 64) {
    float m   = sum[c] / CNT;
    float var = sumsq[c] / CNT - m * m;
    float inv = rsqrtf(var + 1e-5f);
    float aa  = gamma[c] * inv;
    a[c] = aa;
    b[c] = beta[c] - m * aa;
  }
}

// ---- apply BN2 in place on z (NCHW) --------------------------------------
__global__ void bn_apply(float* __restrict__ z, const float* __restrict__ a,
                         const float* __restrict__ b) {
  size_t i = (size_t)blockIdx.x * 256 + threadIdx.x;
  int c = (int)((i >> 14) & 63);        // H*W = 16384 = 2^14
  z[i] = z[i] * a[c] + b[c];
}

extern "C" void kernel_launch(void* const* d_in, const int* in_sizes, int n_in,
                              void* d_out, int out_size, void* d_ws, size_t ws_size,
                              hipStream_t stream) {
  const float* x   = (const float*)d_in[0];
  const float* w1  = (const float*)d_in[1];
  const float* w2  = (const float*)d_in[2];
  const float* g1  = (const float*)d_in[3];
  const float* be1 = (const float*)d_in[4];
  const float* g2  = (const float*)d_in[5];
  const float* be2 = (const float*)d_in[6];
  float* out = (float*)d_out;

  char* ws = (char*)d_ws;
  unsigned short* wbf1 = (unsigned short*)(ws);
  unsigned short* wbf2 = (unsigned short*)(ws + 73728);
  float* stats  = (float*)(ws + 147456);          // sum1,sumsq1,sum2,sumsq2
  float* params = (float*)(ws + 148480);          // a1,b1,a2,b2
  float* btap   = (float*)(ws + 149504);          // 576 floats
  unsigned short* y1 = (unsigned short*)(ws + 151808);

  // 1) repack conv1 weights, zero stats
  prep_w<<<144, 256, 0, stream>>>(w1, wbf1);
  zero_f<<<1, 256, 0, stream>>>(stats, 256);

  dim3 grid(Ww / 32, Hh, Nn);   // (4,128,32)
  // 2) conv1 -> y1 (NHWC bf16) + BN1 stats
  conv_bb<false><<<grid, 256, 0, stream>>>(x, wbf1, nullptr, y1, nullptr,
                                           nullptr, stats, stats + 64);
  // 3) BN1 affine
  bn_params<<<1, 64, 0, stream>>>(stats, stats + 64, g1, be1, params, params + 64);
  // 4) fold BN1 into conv2 weights + per-tap bias
  fold_w2<<<9, 64, 0, stream>>>(w2, params, params + 64, wbf2, btap);
  // 5) conv2 over y1 (async LDS staging) + bias + residual -> z, BN2 stats
  conv_bb<true><<<grid, 256, 0, stream>>>(x, wbf2, y1, nullptr, out,
                                          btap, stats + 128, stats + 192);
  // 6) BN2 affine
  bn_params<<<1, 64, 0, stream>>>(stats + 128, stats + 192, g2, be2,
                                  params + 128, params + 192);
  // 7) normalize z in place
  bn_apply<<<131072, 256, 0, stream>>>(out, params + 128, params + 192);
}
